// MemorySelfAttention_55611236549057
// MI455X (gfx1250) — matrix operand order repaired and checked by reading, not compiled, not run on hardware
//
#include <hip/hip_runtime.h>

#define HEADS  16
#define DIMF   1024
#define DIM_H  64
#define SEQ    512
#define MEML   512
#define CMEML  128
#define KVLEN  1152   // CMEM_LEN + MEM_LEN + SEQ_LEN
#define BATCH  8

typedef __attribute__((ext_vector_type(16))) __bf16 bf16x16;
typedef __attribute__((ext_vector_type(8)))  float  f32x8;

// ---------------------------------------------------------------------------
// Generic batched GEMM on the WMMA pipe, 32x32 C-tile per wave (2x2 subtiles).
//   C[M,N] (f32) = alpha * A[M,K] (f32) x B (f32)  [+ bias[n]]
//   BT=1: B is [N,K] row-major ("NT"), BT=0: B is [K,N] ("NN") - template
//   parameter so each instantiation's K-loop is branch-free and low-pressure.
//   Four v_wmma_f32_16x16x32_bf16 per K-step; f32 operands converted to bf16
//   in-register at fragment load (avoids a separate conversion pass).
//   Batch z: ptr += (z/hz)*Outer + (z%hz)*Inner  (covers (b,h) double index).
// ---------------------------------------------------------------------------
template <int BT>
__global__ __launch_bounds__(32)
void k_wmma_gemm(const float* __restrict__ A, const float* __restrict__ B,
                 float* __restrict__ C, const float* __restrict__ bias,
                 int lda, int ldb, int ldc, int K, float alpha,
                 int hz, long long aO, long long aI, long long bO, long long bI,
                 long long cO, long long cI)
{
    const int z = blockIdx.z;
    A += (long long)(z / hz) * aO + (long long)(z % hz) * aI;
    B += (long long)(z / hz) * bO + (long long)(z % hz) * bI;
    C += (long long)(z / hz) * cO + (long long)(z % hz) * cI;

    const int lane = threadIdx.x;        // blockDim.x == 32: one wave per block
    const int tn   = blockIdx.x * 32;
    const int tm   = blockIdx.y * 32;
    const int l15  = lane & 15;
    const int hi   = (lane >> 4) & 1;    // lane half selects K sub-range

    f32x8 acc00 = {0.f,0.f,0.f,0.f,0.f,0.f,0.f,0.f};
    f32x8 acc01 = acc00, acc10 = acc00, acc11 = acc00;

    const float* arow0 = A + (long long)(tm + l15) * lda;
    const float* arow1 = arow0 + (long long)16 * lda;

    for (int k0 = 0; k0 < K; k0 += 32) {
        bf16x16 a0, a1, b0, b1;
        // A 16x32 bf16 fragment: lanes<16 hold K 0..15, lanes>=16 K 16..31
        {
            const float* ap0 = arow0 + k0 + hi * 16;
            const float* ap1 = arow1 + k0 + hi * 16;
#pragma unroll
            for (int i = 0; i < 8; ++i) {
                a0[i]     = (__bf16)ap0[i];
                a0[i + 8] = (__bf16)ap0[i + 8];
                a1[i]     = (__bf16)ap1[i];
                a1[i + 8] = (__bf16)ap1[i + 8];
            }
            if (k0 + 32 < K) {           // pull next K-block into the near cache
                __builtin_prefetch(ap0 + 32, 0, 3);
                __builtin_prefetch(ap1 + 32, 0, 3);
            }
        }
        // B 32x16 bf16 fragments: N = lane%16, lanes<16 hold K 0..15, lanes>=16 K 16..31
        if (BT) {
            const float* bp0 = B + (long long)(tn + l15) * ldb + k0 + hi * 16;
            const float* bp1 = bp0 + (long long)16 * ldb;
#pragma unroll
            for (int i = 0; i < 16; ++i) {
                b0[i] = (__bf16)bp0[i];
                b1[i] = (__bf16)bp1[i];
            }
        } else {
            const float* bp = B + (long long)(k0 + hi * 16) * ldb + tn + l15;
#pragma unroll
            for (int i = 0; i < 16; ++i) {
                b0[i] = (__bf16)bp[(long long)i * ldb];
                b1[i] = (__bf16)bp[(long long)i * ldb + 16];
            }
        }
        acc00 = __builtin_amdgcn_wmma_f32_16x16x32_bf16(false, a0, false, b0, (short)0, acc00, false, false);
        acc01 = __builtin_amdgcn_wmma_f32_16x16x32_bf16(false, a0, false, b1, (short)0, acc01, false, false);
        acc10 = __builtin_amdgcn_wmma_f32_16x16x32_bf16(false, a1, false, b0, (short)0, acc10, false, false);
        acc11 = __builtin_amdgcn_wmma_f32_16x16x32_bf16(false, a1, false, b1, (short)0, acc11, false, false);
    }

    // C/D layout: VGPR r -> M = tileM + r + (lane>=16 ? 8 : 0), N = tileN + lane%16
    const int   gn0 = tn + l15;
    const int   gn1 = gn0 + 16;
    const float bv0 = bias ? bias[gn0] : 0.0f;
    const float bv1 = bias ? bias[gn1] : 0.0f;
    float* c0 = C + (long long)(tm + hi * 8) * ldc;
    float* c1 = C + (long long)(tm + 16 + hi * 8) * ldc;
#pragma unroll
    for (int r = 0; r < 8; ++r) {
        c0[(long long)r * ldc + gn0] = alpha * acc00[r] + bv0;
        c0[(long long)r * ldc + gn1] = alpha * acc01[r] + bv1;
        c1[(long long)r * ldc + gn0] = alpha * acc10[r] + bv0;
        c1[(long long)r * ldc + gn1] = alpha * acc11[r] + bv1;
    }
}

// kv_input = concat(cmem, mem, x) along seq: (b, 1152, 1024)
__global__ void k_build_kvi(const float* __restrict__ cmem, const float* __restrict__ mem,
                            const float* __restrict__ x, float* __restrict__ kvi)
{
    long long idx = (long long)blockIdx.x * blockDim.x + threadIdx.x;
    const long long total = (long long)BATCH * KVLEN * DIMF;
    if (idx >= total) return;
    int i = (int)(idx % DIMF);
    long long r = idx / DIMF;
    int j = (int)(r % KVLEN);
    int b = (int)(r / KVLEN);
    float v;
    if (j < CMEML)              v = cmem[((long long)b * CMEML + j) * DIMF + i];
    else if (j < CMEML + MEML)  v = mem [((long long)b * MEML + (j - CMEML)) * DIMF + i];
    else                        v = x   [((long long)b * SEQ  + (j - CMEML - MEML)) * DIMF + i];
    kvi[idx] = v;
}

// weights[bi,h,m,n] = scale * (qk[bi,h,m,n] + pos[h,m, n+SEQ-1-m])  (zero past KVLEN)
__global__ void k_combine_shift(float* __restrict__ w, const float* __restrict__ pos,
                                float scale, int bi)
{
    long long idx = (long long)blockIdx.x * blockDim.x + threadIdx.x;
    const long long total = (long long)HEADS * SEQ * KVLEN;
    if (idx >= total) return;
    int n = (int)(idx % KVLEN);
    long long r = idx / KVLEN;
    int m = (int)(r % SEQ);
    int h = (int)(r / SEQ);
    int src = n + (SEQ - 1) - m;                       // relative-position shift gather
    float pv = (src < KVLEN) ? pos[((long long)h * SEQ + m) * KVLEN + src] : 0.0f;
    long long o = (((long long)bi * HEADS + h) * SEQ + m) * KVLEN + n;
    w[o] = scale * (w[o] + pv);
}

// In-place row softmax; one 256-thread block per row.
__global__ void k_softmax_rows(float* __restrict__ p, int cols, int rowStride)
{
    float* row = p + (long long)blockIdx.x * rowStride;
    __shared__ float s[256];
    const int t = threadIdx.x;
    float m = -1e30f;
    for (int c = t; c < cols; c += 256) m = fmaxf(m, row[c]);
    s[t] = m; __syncthreads();
    for (int o = 128; o > 0; o >>= 1) { if (t < o) s[t] = fmaxf(s[t], s[t + o]); __syncthreads(); }
    const float mx = s[0]; __syncthreads();
    float sum = 0.f;
    for (int c = t; c < cols; c += 256) { float e = __expf(row[c] - mx); row[c] = e; sum += e; }
    s[t] = sum; __syncthreads();
    for (int o = 128; o > 0; o >>= 1) { if (t < o) s[t] += s[t + o]; __syncthreads(); }
    const float inv = 1.0f / s[0];
    for (int c = t; c < cols; c += 256) row[c] *= inv;
}

// Repack conv_w (o,i,r) -> cwt[o, r*1024+i] so conv_compress becomes a plain NT GEMM.
__global__ void k_repack_convw(const float* __restrict__ w, float* __restrict__ out)
{
    long long idx = (long long)blockIdx.x * blockDim.x + threadIdx.x;
    const long long total = (long long)DIMF * DIMF * 4;
    if (idx >= total) return;
    int kk = (int)(idx % (DIMF * 4));
    int o  = (int)(idx / (DIMF * 4));
    int r  = kk / DIMF;
    int i  = kk % DIMF;
    out[idx] = w[(long long)o * (DIMF * 4) + i * 4 + r];
}

__global__ void k_sqdiff_accum(const float* __restrict__ a, const float* __restrict__ b,
                               int n, float* __restrict__ accum)
{
    __shared__ float s[256];
    const int t = threadIdx.x;
    long long idx = (long long)blockIdx.x * blockDim.x + t;
    float d = 0.f;
    if (idx < n) { float x = a[idx] - b[idx]; d = x * x; }
    s[t] = d; __syncthreads();
    for (int o = 128; o > 0; o >>= 1) { if (t < o) s[t] += s[t + o]; __syncthreads(); }
    if (t == 0) atomicAdd(accum, s[0]);
}

__global__ void k_zero(float* p)                                    { p[0] = 0.0f; }
__global__ void k_finalize(const float* a, float* d, float inv)     { d[0] = a[0] * inv; }

// ---------------------------------------------------------------------------
extern "C" void kernel_launch(void* const* d_in, const int* in_sizes, int n_in,
                              void* d_out, int out_size, void* d_ws, size_t ws_size,
                              hipStream_t stream)
{
    const float* x     = (const float*)d_in[0];
    const float* mem   = (const float*)d_in[1];
    const float* cmem  = (const float*)d_in[2];
    const float* pe    = (const float*)d_in[3];
    const float* Wq    = (const float*)d_in[4];
    const float* Wkv   = (const float*)d_in[5];
    const float* Wout  = (const float*)d_in[6];
    const float* bout  = (const float*)d_in[7];
    const float* convw = (const float*)d_in[8];
    const float* convb = (const float*)d_in[9];

    float* out = (float*)d_out;
    // Output tuple layout (flat, f32): logits | new_mem | new_cmem | aux_loss | weights
    float* logits  = out;                    // 8*512*1024
    float* newmem  = out + 4194304LL;        // 8*512*1024
    float* newcm   = out + 8388608LL;        // 8*128*1024
    float* auxl    = out + 9437184LL;        // 1
    float* weights = out + 9437185LL;        // 8*16*512*1152

    float* ws = (float*)d_ws;                // ~235 MB used
    float* kvi  = ws;                        //  9,437,184  kv_input (b,1152,1024)
    float* kv   = ws + 9437184LL;            // 18,874,368  [k|v]    (b*1152,2048)
    float* q    = ws + 28311552LL;           //  4,194,304  q        (b*512,1024)
    float* pos  = ws + 32505856LL;           //  9,437,184  q·peT, one batch (h,512,1152)
    float* obuf = ws + 41943040LL;           //  4,194,304  attn·v   (b,512,1024)
    float* cmkv = ws + 46137344LL;           //  2,097,152  cmem k|v (b*128,2048)
    float* cwt  = ws + 48234496LL;           //  4,194,304  repacked conv weight
    float* ad1  = ws + 52428800LL;           //  4,194,304  aux dots1, one batch
    float* ad2  = ws + 56623104LL;           //  1,048,576  aux dots2, one batch
    float* ao1  = ws + 57671680LL;           //    524,288  aux out1
    float* ao2  = ws + 58195968LL;           //    524,288  aux out2
    float* acc  = ws + 58720256LL;           //          1  aux accumulator

    const float scale = 0.125f;              // DIM_H^-0.5

    auto gemm = [&](const float* A, int lda, const float* B, int ldb, float* C, int ldc,
                    int M, int N, int K, bool bt, float alpha, const float* bias,
                    int batches, int hz,
                    long long aO, long long aI, long long bO, long long bI,
                    long long cO, long long cI) {
        dim3 g((unsigned)(N / 32), (unsigned)(M / 32), (unsigned)batches), blk(32);
        if (bt)
            k_wmma_gemm<1><<<g, blk, 0, stream>>>(A, B, C, bias, lda, ldb, ldc, K,
                                                  alpha, hz, aO, aI, bO, bI, cO, cI);
        else
            k_wmma_gemm<0><<<g, blk, 0, stream>>>(A, B, C, bias, lda, ldb, ldc, K,
                                                  alpha, hz, aO, aI, bO, bI, cO, cI);
    };

    // 1) kv_input = concat(cmem, mem, x);  new_mem = x
    {
        long long total = (long long)BATCH * KVLEN * DIMF;
        k_build_kvi<<<(unsigned)((total + 255) / 256), 256, 0, stream>>>(cmem, mem, x, kvi);
        hipMemcpyAsync(newmem, x, 4194304LL * sizeof(float), hipMemcpyDeviceToDevice, stream);
    }

    // 2) q = x @ Wq        (4096 x 1024 x 1024)
    gemm(x, DIMF, Wq, DIMF, q, DIMF, BATCH * SEQ, DIMF, DIMF, false, 1.f, nullptr,
         1, 1, 0, 0, 0, 0, 0, 0);
    // 3) [k|v] = kv_input @ Wkv   (9216 x 2048 x 1024)
    gemm(kvi, DIMF, Wkv, 2 * DIMF, kv, 2 * DIMF, BATCH * KVLEN, 2 * DIMF, DIMF, false, 1.f,
         nullptr, 1, 1, 0, 0, 0, 0, 0, 0);

    // 4) dots = q · kT, per (b,h): 512 x 1152 x 64, NT   -> weights (pre-scale)
    gemm(q, DIMF, kv, 2 * DIMF, weights, KVLEN, SEQ, KVLEN, DIM_H, true, 1.f, nullptr,
         BATCH * HEADS, HEADS,
         (long long)SEQ * DIMF, DIM_H,
         (long long)KVLEN * 2 * DIMF, DIM_H,
         (long long)HEADS * SEQ * KVLEN, (long long)SEQ * KVLEN);

    // 5) per batch: pos_dots = q · peT (per head), then fused shift + scale + add
    for (int bi = 0; bi < BATCH; ++bi) {
        gemm(q + (long long)bi * SEQ * DIMF, DIMF, pe, DIM_H, pos, KVLEN,
             SEQ, KVLEN, DIM_H, true, 1.f, nullptr, HEADS, HEADS,
             0, DIM_H, 0, (long long)KVLEN * DIM_H, 0, (long long)SEQ * KVLEN);
        long long total = (long long)HEADS * SEQ * KVLEN;
        k_combine_shift<<<(unsigned)((total + 255) / 256), 256, 0, stream>>>(weights, pos, scale, bi);
    }

    // 6) softmax over kv dim -> attention weights (output tensor, reused as A below)
    k_softmax_rows<<<(unsigned)(BATCH * HEADS * SEQ), 256, 0, stream>>>(weights, KVLEN, KVLEN);

    // 7) out = attn @ v, per (b,h): 512 x 64 x 1152, NN (v = cols 1024.. of kv)
    gemm(weights, KVLEN, kv + DIMF, 2 * DIMF, obuf, DIMF, SEQ, DIM_H, KVLEN, false, 1.f,
         nullptr, BATCH * HEADS, HEADS,
         (long long)HEADS * SEQ * KVLEN, (long long)SEQ * KVLEN,
         (long long)KVLEN * 2 * DIMF, DIM_H,
         (long long)SEQ * DIMF, DIM_H);

    // 8) logits = out @ Wout + bout
    gemm(obuf, DIMF, Wout, DIMF, logits, DIMF, BATCH * SEQ, DIMF, DIMF, false, 1.f, bout,
         1, 1, 0, 0, 0, 0, 0, 0);

    // 9) conv_compress(mem) as GEMM: mem viewed (b*128, 4096) x cwt(1024,4096)^T + conv_b
    {
        long long total = (long long)DIMF * DIMF * 4;
        k_repack_convw<<<(unsigned)((total + 255) / 256), 256, 0, stream>>>(convw, cwt);
        gemm(mem, DIMF * 4, cwt, DIMF * 4, newcm, DIMF, BATCH * CMEML, DIMF, DIMF * 4,
             true, 1.f, convb, 1, 1, 0, 0, 0, 0, 0, 0);
    }

    // 10) cmem_k|v = compressed @ Wkv  (1024 x 2048 x 1024)
    gemm(newcm, DIMF, Wkv, 2 * DIMF, cmkv, 2 * DIMF, BATCH * CMEML, 2 * DIMF, DIMF, false,
         1.f, nullptr, 1, 1, 0, 0, 0, 0, 0, 0);

    // 11) aux loss: || attn(q, old_k, old_v) - attn(q, cmem_k, cmem_v) ||^2 mean
    k_zero<<<1, 1, 0, stream>>>(acc);
    for (int bi = 0; bi < BATCH; ++bi) {
        const float* qb = q + (long long)bi * SEQ * DIMF;
        const float* okv = kv + ((long long)bi * KVLEN + CMEML) * 2 * DIMF;  // j in [128,640)
        // dots1 = scale * q · old_kT  (512 x 512 x 64)
        gemm(qb, DIMF, okv, 2 * DIMF, ad1, MEML, SEQ, MEML, DIM_H, true, scale, nullptr,
             HEADS, HEADS, 0, DIM_H, 0, DIM_H, 0, (long long)SEQ * MEML);
        k_softmax_rows<<<(unsigned)(HEADS * SEQ), 256, 0, stream>>>(ad1, MEML, MEML);
        // out1 = attn1 @ old_v   (512 x 64 x 512)
        gemm(ad1, MEML, okv + DIMF, 2 * DIMF, ao1, DIM_H, SEQ, DIM_H, MEML, false, 1.f,
             nullptr, HEADS, HEADS, 0, (long long)SEQ * MEML, 0, DIM_H, 0, (long long)SEQ * DIM_H);
        // dots2 = scale * q · cmem_kT  (512 x 128 x 64)
        const float* ckv = cmkv + (long long)bi * CMEML * 2 * DIMF;
        gemm(qb, DIMF, ckv, 2 * DIMF, ad2, CMEML, SEQ, CMEML, DIM_H, true, scale, nullptr,
             HEADS, HEADS, 0, DIM_H, 0, DIM_H, 0, (long long)SEQ * CMEML);
        k_softmax_rows<<<(unsigned)(HEADS * SEQ), 256, 0, stream>>>(ad2, CMEML, CMEML);
        // out2 = attn2 @ cmem_v  (512 x 64 x 128)
        gemm(ad2, CMEML, ckv + DIMF, 2 * DIMF, ao2, DIM_H, SEQ, DIM_H, CMEML, false, 1.f,
             nullptr, HEADS, HEADS, 0, (long long)SEQ * CMEML, 0, DIM_H, 0, (long long)SEQ * DIM_H);
        // accumulate squared difference
        k_sqdiff_accum<<<(unsigned)((HEADS * SEQ * DIM_H + 255) / 256), 256, 0, stream>>>(
            ao1, ao2, HEADS * SEQ * DIM_H, acc);
    }
    k_finalize<<<1, 1, 0, stream>>>(acc, auxl, 1.0f / (float)(BATCH * HEADS * SEQ * DIM_H));

    (void)in_sizes; (void)n_in; (void)out_size; (void)ws_size;
}
